// SREA_69226282877207
// MI455X (gfx1250) — matrix operand order repaired and checked by reading, not compiled, not run on hardware
//
#include <hip/hip_runtime.h>
#include <cstdint>
#include <cstddef>

#define N_ENT   100000
#define E_TRI   400000
#define EH2     800000   // 2*EH symmetric edge count
#define R_REL   2000
#define RHID    100
#define EHID    300
#define LGE     60000
#define NEG_SLOPE 0.01f
#define TPB 256

typedef __attribute__((ext_vector_type(16))) __bf16    v16bf;
typedef __attribute__((ext_vector_type(8)))  float     v8f;
typedef __attribute__((ext_vector_type(8)))  unsigned  v8u;

// ---------------- helpers ----------------
__device__ __forceinline__ unsigned pk2bf(float lo, float hi) {
  // pack two fp32 into two bf16 (truncate): low half = lo, high half = hi
  return (__builtin_bit_cast(unsigned, hi) & 0xFFFF0000u) |
         (__builtin_bit_cast(unsigned, lo) >> 16);
}
__device__ __forceinline__ unsigned fenc(float f) {   // order-preserving float->u32
  unsigned u = __builtin_bit_cast(unsigned, f);
  return (u & 0x80000000u) ? ~u : (u | 0x80000000u);
}
__device__ __forceinline__ float fdec(unsigned u) {
  unsigned b = (u & 0x80000000u) ? (u & 0x7FFFFFFFu) : ~u;
  return __builtin_bit_cast(float, b);
}
__device__ __forceinline__ float leaky(float x) { return x > 0.f ? x : NEG_SLOPE * x; }
__device__ __forceinline__ float4 ld4(const float* p) {
  return *reinterpret_cast<const float4*>(p);
}

// Build a v16bf A-fragment from two contiguous 8-float runs (r0 -> elems 0..7, r1 -> 8..15)
__device__ __forceinline__ v16bf mkfrag(float4 a, float4 b, float4 c, float4 d) {
  v8u u;
  u[0] = pk2bf(a.x, a.y); u[1] = pk2bf(a.z, a.w);
  u[2] = pk2bf(b.x, b.y); u[3] = pk2bf(b.z, b.w);
  u[4] = pk2bf(c.x, c.y); u[5] = pk2bf(c.z, c.w);
  u[6] = pk2bf(d.x, d.y); u[7] = pk2bf(d.z, d.w);
  return __builtin_bit_cast(v16bf, u);
}

// ---------------- WMMA GEMM: C[M,N] = A[M,K] * W[N,K]^T (bf16 in, f32 acc) ----------------
// One wave computes a 32x32 tile of C (2x2 of 16x16 WMMA tiles); 8 waves/block -> 256 rows.
__global__ void srea_gemm_bf16(const float* __restrict__ A, int lda,
                               const float* __restrict__ W, int ldw,
                               float* __restrict__ C, int ldc,
                               int M, int K, int N)
{
  const int wave = threadIdx.x >> 5;
  const int lane = threadIdx.x & 31;
  const int half = lane >> 4;          // 0 or 1 (lane group)
  const int l16  = lane & 15;
  const int m0 = (blockIdx.y * 8 + wave) * 32;
  const int n0 = blockIdx.x * 32;
  if (m0 >= M || n0 >= N) return;      // wave-uniform exit, EXEC stays all-ones for WMMA

  const int mA = m0 + l16, mB = m0 + 16 + l16;
  const int nA = n0 + l16, nB = n0 + 16 + l16;
  const bool mokA = (mA < M), mokB = (mB < M);
  const bool nokA = (nA < N), nokB = (nB < N);
  const float* arow0 = A + (size_t)(mokA ? mA : (M - 1)) * lda;
  const float* arow1 = A + (size_t)(mokB ? mB : (M - 1)) * lda;
  const float* wrow0 = W + (size_t)(nokA ? nA : (N - 1)) * ldw;
  const float* wrow1 = W + (size_t)(nokB ? nB : (N - 1)) * ldw;

  v8f acc00 = {}, acc01 = {}, acc10 = {}, acc11 = {};

  const int Kfull = (K / 32) * 32;
  const int aoff = 8 * half;           // A: lanes 16-31 take K+8 / K+24 runs
  const int boff = 8 * half;           // B: same runs as A (K+8 / K+24 for lanes 16-31)

  for (int kc = 0; kc < Kfull; kc += 32) {
    // A fragments: two contiguous 8-float runs per lane -> 4 x b128 loads per row-set
    v16bf af0 = mkfrag(ld4(arow0 + kc + aoff),      ld4(arow0 + kc + aoff + 4),
                       ld4(arow0 + kc + 16 + aoff), ld4(arow0 + kc + 16 + aoff + 4));
    v16bf af1 = mkfrag(ld4(arow1 + kc + aoff),      ld4(arow1 + kc + aoff + 4),
                       ld4(arow1 + kc + 16 + aoff), ld4(arow1 + kc + 16 + aoff + 4));
    // B fragments: two contiguous 8-float runs per lane (same k order as A)
    v16bf bf0 = mkfrag(ld4(wrow0 + kc + boff),      ld4(wrow0 + kc + boff + 4),
                       ld4(wrow0 + kc + 16 + boff), ld4(wrow0 + kc + 16 + boff + 4));
    v16bf bf1 = mkfrag(ld4(wrow1 + kc + boff),      ld4(wrow1 + kc + boff + 4),
                       ld4(wrow1 + kc + 16 + boff), ld4(wrow1 + kc + 16 + boff + 4));

    acc00 = __builtin_amdgcn_wmma_f32_16x16x32_bf16(false, af0, false, bf0, (short)0, acc00, false, false);
    acc01 = __builtin_amdgcn_wmma_f32_16x16x32_bf16(false, af0, false, bf1, (short)0, acc01, false, false);
    acc10 = __builtin_amdgcn_wmma_f32_16x16x32_bf16(false, af1, false, bf0, (short)0, acc10, false, false);
    acc11 = __builtin_amdgcn_wmma_f32_16x16x32_bf16(false, af1, false, bf1, (short)0, acc11, false, false);
  }

  if (Kfull < K) {                      // tail chunk: clamp+mask, branch-free loads
    const int kc = Kfull;
    v8u ua0, ua1, ub0, ub1;
#pragma unroll
    for (int w = 0; w < 8; ++w) {
      // A element w*2+p lives at run r = w/4 (0: +aoff, 1: +16+aoff), pos (w%4)*2+p
      const int ka = kc + (w >> 2) * 16 + aoff + (w & 3) * 2;
      float a0lo = arow0[(ka     < K) ? ka     : (K - 1)] * ((ka     < K) ? 1.f : 0.f);
      float a0hi = arow0[(ka + 1 < K) ? ka + 1 : (K - 1)] * ((ka + 1 < K) ? 1.f : 0.f);
      float a1lo = arow1[(ka     < K) ? ka     : (K - 1)] * ((ka     < K) ? 1.f : 0.f);
      float a1hi = arow1[(ka + 1 < K) ? ka + 1 : (K - 1)] * ((ka + 1 < K) ? 1.f : 0.f);
      ua0[w] = pk2bf(a0lo, a0hi);
      ua1[w] = pk2bf(a1lo, a1hi);
      const int kb = kc + (w >> 2) * 16 + boff + (w & 3) * 2;
      float b0lo = wrow0[(kb     < K) ? kb     : (K - 1)] * ((kb     < K) ? 1.f : 0.f);
      float b0hi = wrow0[(kb + 1 < K) ? kb + 1 : (K - 1)] * ((kb + 1 < K) ? 1.f : 0.f);
      float b1lo = wrow1[(kb     < K) ? kb     : (K - 1)] * ((kb     < K) ? 1.f : 0.f);
      float b1hi = wrow1[(kb + 1 < K) ? kb + 1 : (K - 1)] * ((kb + 1 < K) ? 1.f : 0.f);
      ub0[w] = pk2bf(b0lo, b0hi);
      ub1[w] = pk2bf(b1lo, b1hi);
    }
    v16bf af0 = __builtin_bit_cast(v16bf, ua0);
    v16bf af1 = __builtin_bit_cast(v16bf, ua1);
    v16bf bf0 = __builtin_bit_cast(v16bf, ub0);
    v16bf bf1 = __builtin_bit_cast(v16bf, ub1);
    acc00 = __builtin_amdgcn_wmma_f32_16x16x32_bf16(false, af0, false, bf0, (short)0, acc00, false, false);
    acc01 = __builtin_amdgcn_wmma_f32_16x16x32_bf16(false, af0, false, bf1, (short)0, acc01, false, false);
    acc10 = __builtin_amdgcn_wmma_f32_16x16x32_bf16(false, af1, false, bf0, (short)0, acc10, false, false);
    acc11 = __builtin_amdgcn_wmma_f32_16x16x32_bf16(false, af1, false, bf1, (short)0, acc11, false, false);
  }

  // C/D layout: lane l16 = N, half selects M block of 8, VGPR v -> M = 8*half + v
#pragma unroll
  for (int v = 0; v < 8; ++v) {
    const int mr0 = m0 + half * 8 + v;
    const int mr1 = mr0 + 16;
    if (mr0 < M && nokA) C[(size_t)mr0 * ldc + nA] = acc00[v];
    if (mr0 < M && nokB) C[(size_t)mr0 * ldc + nB] = acc01[v];
    if (mr1 < M && nokA) C[(size_t)mr1 * ldc + nA] = acc10[v];
    if (mr1 < M && nokB) C[(size_t)mr1 * ldc + nB] = acc11[v];
  }
}

// ---------------- elementwise / utility ----------------
__global__ void srea_zero(float* __restrict__ p, long n) {
  long i = (long)blockIdx.x * blockDim.x + threadIdx.x;
  if (i < n) p[i] = 0.0f;
}
__global__ void srea_fill_u32(unsigned* __restrict__ p, unsigned v, long n) {
  long i = (long)blockIdx.x * blockDim.x + threadIdx.x;
  if (i < n) p[i] = v;
}
__global__ void srea_add(const float* __restrict__ a, const float* __restrict__ b,
                         float* __restrict__ o, long n) {
  long i = (long)blockIdx.x * blockDim.x + threadIdx.x;
  if (i < n) o[i] = a[i] + b[i];
}
__global__ void srea_relu(float* __restrict__ p, long n) {
  long i = (long)blockIdx.x * blockDim.x + threadIdx.x;
  if (i < n) { float v = p[i]; p[i] = v > 0.f ? v : 0.f; }
}
__global__ void srea_relu2d(float* __restrict__ p, int ld, long rows, int cols) {
  long i = (long)blockIdx.x * blockDim.x + threadIdx.x;
  if (i < rows * (long)cols) {
    long r = i / cols; int c = (int)(i - r * cols);
    float v = p[r * ld + c]; p[r * ld + c] = v > 0.f ? v : 0.f;
  }
}

// ---------------- degree / norm ----------------
__global__ void srea_deg(const int* __restrict__ dst, float* __restrict__ deg, int E) {
  int e = blockIdx.x * blockDim.x + threadIdx.x;
  if (e < E) atomicAdd(&deg[dst[e]], 1.0f);
}
__global__ void srea_dinv(const float* __restrict__ deg, float* __restrict__ dinv, int n) {
  int i = blockIdx.x * blockDim.x + threadIdx.x;
  if (i < n) { float d = deg[i]; dinv[i] = d > 0.f ? rsqrtf(d) : 0.f; }
}

// ---------------- GCN scatter: agg[i] += dinv[j]*dinv[i]*xw[j]  (wave per edge) ----------------
__global__ void srea_gcn_scatter(const float* __restrict__ xw,
                                 const int* __restrict__ src, const int* __restrict__ dst,
                                 const float* __restrict__ dinv,
                                 float* __restrict__ agg, int F, int E)
{
  int e = blockIdx.x * (blockDim.x >> 5) + (threadIdx.x >> 5);
  int lane = threadIdx.x & 31;
  if (e >= E) return;
  int j = src[e], i = dst[e];
  float norm = dinv[j] * dinv[i];
  const float* xr = xw + (size_t)j * F;
  float* o = agg + (size_t)i * F;
  __builtin_prefetch(xr, 0, 1);
  for (int f = lane; f < F; f += 32) atomicAdd(&o[f], norm * xr[f]);
}

// ---------------- highway: out = sig(lin+b) * relu(agg) + (1-sig) * x_in ----------------
__global__ void srea_highway(const float* __restrict__ x_in, int ld_in,
                             const float* __restrict__ agg, int ld_agg,
                             const float* __restrict__ lin, int ld_lin,
                             const float* __restrict__ bias,
                             float* __restrict__ outp, int ld_out, long M, int F)
{
  long i = (long)blockIdx.x * blockDim.x + threadIdx.x;
  if (i >= M * (long)F) return;
  long r = i / F; int c = (int)(i - r * F);
  float g = 1.0f / (1.0f + __expf(-(lin[r * ld_lin + c] + bias[c])));
  float x2 = agg[r * ld_agg + c]; x2 = x2 > 0.f ? x2 : 0.f;
  float x1 = x_in[r * ld_in + c];
  outp[r * ld_out + c] = g * x2 + (1.0f - g) * x1;
}

// ---------------- segment softmax primitives (unsorted indices) ----------------
__global__ void srea_smax(const float* __restrict__ val, const int* __restrict__ idx,
                          unsigned* __restrict__ m, int E) {
  int e = blockIdx.x * blockDim.x + threadIdx.x;
  if (e < E) atomicMax(&m[idx[e]], fenc(val[e]));
}
__global__ void srea_sexp(const float* __restrict__ val, const int* __restrict__ idx,
                          const unsigned* __restrict__ m, float* __restrict__ p,
                          float* __restrict__ s, int E) {
  int e = blockIdx.x * blockDim.x + threadIdx.x;
  if (e < E) {
    float pe = __expf(val[e] - fdec(m[idx[e]]));
    p[e] = pe;
    atomicAdd(&s[idx[e]], pe);
  }
}
__global__ void srea_snorm(float* __restrict__ p, const int* __restrict__ idx,
                           const float* __restrict__ s, int E) {
  int e = blockIdx.x * blockDim.x + threadIdx.x;
  if (e < E) p[e] = p[e] / (s[idx[e]] + 1e-16f);
}

// ---------------- attention scatter: out[dst] += alpha * xsrc[srcidx]  (wave per edge) ---------
__global__ void srea_attn_scatter(const float* __restrict__ xsrc, int ldsrc,
                                  const int* __restrict__ srcidx,
                                  const int* __restrict__ dstidx,
                                  const float* __restrict__ alpha,
                                  float* __restrict__ outp, int ldout, int F, int E)
{
  int e = blockIdx.x * (blockDim.x >> 5) + (threadIdx.x >> 5);
  int lane = threadIdx.x & 31;
  if (e >= E) return;
  float a = alpha[e];
  const float* xr = xsrc + (size_t)srcidx[e] * ldsrc;
  float* o = outp + (size_t)dstidx[e] * ldout;
  __builtin_prefetch(xr, 0, 1);
  for (int f = lane; f < F; f += 32) atomicAdd(&o[f], a * xr[f]);
}

// ---------------- row dot: out[r] (+)= X[r,:] . a  (wave per row, shuffle reduce) --------------
__global__ void srea_rowdot(const float* __restrict__ X, int ldx,
                            const float* __restrict__ a,
                            float* __restrict__ outp, int rows, int F, int accum)
{
  int w = blockIdx.x * (blockDim.x >> 5) + (threadIdx.x >> 5);
  int lane = threadIdx.x & 31;
  if (w >= rows) return;
  const float* row = X + (size_t)w * ldx;
  float s = 0.f;
  for (int f = lane; f < F; f += 32) s += row[f] * a[f];
#pragma unroll
  for (int o = 16; o > 0; o >>= 1) s += __shfl_xor(s, o, 32);
  if (lane == 0) outp[w] = accum ? (outp[w] + s) : s;
}

// ---------------- GAT edge scores ----------------
__global__ void srea_gat_e(const int* __restrict__ src, const int* __restrict__ dst,
                           const int* __restrict__ rel_all,
                           const float* __restrict__ aip, const float* __restrict__ ajp,
                           const float* __restrict__ rpj, float* __restrict__ e, int E, int R)
{
  int t = blockIdx.x * blockDim.x + threadIdx.x;
  if (t >= E) return;
  int rr = rel_all[t] % R;   // rel_emb = vstack([rel_cat, rel_cat])
  e[t] = leaky(aip[dst[t]] + ajp[src[t]] + rpj[rr]);
}
__global__ void srea_g2e_e(const int* __restrict__ node, const int* __restrict__ rel,
                           const float* __restrict__ nproj, const float* __restrict__ arp,
                           float* __restrict__ e, int E)
{
  int t = blockIdx.x * blockDim.x + threadIdx.x;
  if (t >= E) return;
  e[t] = leaky(nproj[node[t]] + arp[rel[t]]);
}

// ---------------- host ----------------
static inline int gsz(long n, int per) { return (int)((n + per - 1) / per); }

extern "C" void kernel_launch(void* const* d_in, const int* in_sizes, int n_in,
                              void* d_out_, int out_size, void* d_ws, size_t ws_size,
                              hipStream_t stream)
{
  (void)in_sizes; (void)n_in; (void)out_size; (void)ws_size;
  const float* x_e          = (const float*)d_in[0];
  const float* merge_val    = (const float*)d_in[1];
  const float* tri_val      = (const float*)d_in[2];
  const float* gcn1_w       = (const float*)d_in[3];
  const float* hw1_w        = (const float*)d_in[4];
  const float* hw1_b        = (const float*)d_in[5];
  const float* gcn2_w       = (const float*)d_in[6];
  const float* hw2_w        = (const float*)d_in[7];
  const float* hw2_b        = (const float*)d_in[8];
  const float* rel_emb_out1 = (const float*)d_in[9];
  const float* rel_emb_tri1 = (const float*)d_in[10];
  const float* hwr_w        = (const float*)d_in[11];
  const float* hwr_b        = (const float*)d_in[12];
  const float* gat_ai       = (const float*)d_in[13];
  const float* gat_aj       = (const float*)d_in[14];
  const float* gat_ar       = (const float*)d_in[15];
  const float* g2e_ah       = (const float*)d_in[16];
  const float* g2e_at       = (const float*)d_in[17];
  const float* g2e_ar       = (const float*)d_in[18];
  const int* edge_index     = (const int*)d_in[19];   // [2, E_TRI]
  const int* rel            = (const int*)d_in[20];   // [E_TRI]
  const int* ei_all         = (const int*)d_in[21];   // [2, EH2]
  const int* rel_all        = (const int*)d_in[22];   // [EH2]
  const int* lg_merge       = (const int*)d_in[23];   // [2, LGE]
  const int* lg_tri         = (const int*)d_in[24];   // [2, LGE]
  float* out = (float*)d_out_;                         // [N_ENT, 800]

  // ---- workspace carve-up (floats) ----
  float* ws = (float*)d_ws;
  size_t off = 0;
  auto alloc = [&](size_t n) { float* p = ws + off; off += n; return p; };
  float* bufA   = alloc((size_t)N_ENT * EHID);  // gemm out
  float* bufB   = alloc((size_t)N_ENT * EHID);  // gcn aggregate
  float* bufC   = alloc((size_t)N_ENT * EHID);  // x after layer 1
  float* deg    = alloc(N_ENT);
  float* dinv   = alloc(N_ENT);
  float* prA    = alloc(N_ENT);
  float* prB    = alloc(N_ENT);
  unsigned* mbuf = (unsigned*)alloc(N_ENT);
  float* sbuf   = alloc(N_ENT);
  float* ebuf   = alloc(EH2);
  float* ebuf2  = alloc(EH2);
  float* ebuf3  = alloc(EH2);
  float* relM   = alloc((size_t)R_REL * RHID);
  float* relT   = alloc((size_t)R_REL * RHID);
  float* xr     = alloc((size_t)R_REL * RHID);
  float* relLin = alloc((size_t)R_REL * RHID);
  float* rpj    = alloc(R_REL);
  float* arpj   = alloc(R_REL);

  auto gemm = [&](const float* A, int lda, const float* W, int ldw,
                  float* C, int ldc, int M, int K, int N) {
    dim3 g(gsz(N, 32), gsz(M, 256));
    srea_gemm_bf16<<<g, TPB, 0, stream>>>(A, lda, W, ldw, C, ldc, M, K, N);
  };
  auto segsm = [&](const float* val, float* o, const int* idx, int nseg, int E) {
    srea_fill_u32<<<gsz(nseg, TPB), TPB, 0, stream>>>(mbuf, 0x007FFFFFu, nseg); // enc(-inf)
    srea_smax<<<gsz(E, TPB), TPB, 0, stream>>>(val, idx, mbuf, E);
    srea_zero<<<gsz(nseg, TPB), TPB, 0, stream>>>(sbuf, nseg);
    srea_sexp<<<gsz(E, TPB), TPB, 0, stream>>>(val, idx, mbuf, o, sbuf, E);
    srea_snorm<<<gsz(E, TPB), TPB, 0, stream>>>(o, idx, sbuf, E);
  };
  auto lgat = [&](const float* x, const int* lg, const float* val, float* outp) {
    const int* jrow = lg;          // ei[0]
    const int* irow = lg + LGE;    // ei[1]
    segsm(val, ebuf, irow, R_REL, LGE);                  // vi
    segsm(val, ebuf2, jrow, R_REL, LGE);                 // vj
    srea_add<<<gsz(LGE, TPB), TPB, 0, stream>>>(ebuf, ebuf2, ebuf3, LGE);
    segsm(ebuf3, ebuf3, jrow, R_REL, LGE);               // alpha (softmax over j)
    srea_zero<<<gsz((long)R_REL * RHID, TPB), TPB, 0, stream>>>(outp, (long)R_REL * RHID);
    srea_attn_scatter<<<gsz(LGE, 8), TPB, 0, stream>>>(x, RHID, jrow, irow, ebuf3,
                                                       outp, RHID, RHID, LGE);
    srea_relu<<<gsz((long)R_REL * RHID, TPB), TPB, 0, stream>>>(outp, (long)R_REL * RHID);
  };

  // output buffer is poisoned by harness; scatter targets need zeros
  srea_zero<<<gsz((long)N_ENT * 800, TPB), TPB, 0, stream>>>(out, (long)N_ENT * 800);

  // ---- degrees (on edge_index_all targets) ----
  srea_zero<<<gsz(N_ENT, TPB), TPB, 0, stream>>>(deg, N_ENT);
  srea_deg<<<gsz(EH2, TPB), TPB, 0, stream>>>(ei_all + EH2, deg, EH2);
  srea_dinv<<<gsz(N_ENT, TPB), TPB, 0, stream>>>(deg, dinv, N_ENT);

  // ---- GCN + highway, layer 1 ----
  gemm(x_e, EHID, gcn1_w, EHID, bufA, EHID, N_ENT, EHID, EHID);
  srea_zero<<<gsz((long)N_ENT * EHID, TPB), TPB, 0, stream>>>(bufB, (long)N_ENT * EHID);
  srea_gcn_scatter<<<gsz(EH2, 8), TPB, 0, stream>>>(bufA, ei_all, ei_all + EH2, dinv,
                                                    bufB, EHID, EH2);
  gemm(x_e, EHID, hw1_w, EHID, bufA, EHID, N_ENT, EHID, EHID);
  srea_highway<<<gsz((long)N_ENT * EHID, TPB), TPB, 0, stream>>>(
      x_e, EHID, bufB, EHID, bufA, EHID, hw1_b, bufC, EHID, N_ENT, EHID);

  // ---- GCN + highway, layer 2 (writes x2 into out cols 0..299) ----
  gemm(bufC, EHID, gcn2_w, EHID, bufA, EHID, N_ENT, EHID, EHID);
  srea_zero<<<gsz((long)N_ENT * EHID, TPB), TPB, 0, stream>>>(bufB, (long)N_ENT * EHID);
  srea_gcn_scatter<<<gsz(EH2, 8), TPB, 0, stream>>>(bufA, ei_all, ei_all + EH2, dinv,
                                                    bufB, EHID, EH2);
  gemm(bufC, EHID, hw2_w, EHID, bufA, EHID, N_ENT, EHID, EHID);
  srea_highway<<<gsz((long)N_ENT * EHID, TPB), TPB, 0, stream>>>(
      bufC, EHID, bufB, EHID, bufA, EHID, hw2_b, out, 800, N_ENT, EHID);

  // ---- line-graph GATs on relations ----
  lgat(rel_emb_out1, lg_merge, merge_val, relM);   // rel_merge
  lgat(rel_emb_tri1, lg_tri, tri_val, relT);       // rel_tri
  gemm(relM, RHID, hwr_w, RHID, relLin, RHID, R_REL, RHID, RHID);
  srea_highway<<<gsz((long)R_REL * RHID, TPB), TPB, 0, stream>>>(
      relM, RHID, relT, RHID, relLin, RHID, hwr_b, xr, RHID, R_REL, RHID);

  // relproj[r] = rel_cat[r] . gat_ar  (split over the two halves)
  srea_rowdot<<<gsz(R_REL, 8), TPB, 0, stream>>>(relM, RHID, gat_ar, rpj, R_REL, RHID, 0);
  srea_rowdot<<<gsz(R_REL, 8), TPB, 0, stream>>>(relT, RHID, gat_ar + RHID, rpj, R_REL, RHID, 1);

  // ---- relation-aware GAT over all edges (out cols 300..599) ----
  srea_rowdot<<<gsz(N_ENT, 8), TPB, 0, stream>>>(out, 800, gat_ai, prA, N_ENT, EHID, 0);
  srea_rowdot<<<gsz(N_ENT, 8), TPB, 0, stream>>>(out, 800, gat_aj, prB, N_ENT, EHID, 0);
  srea_gat_e<<<gsz(EH2, TPB), TPB, 0, stream>>>(ei_all, ei_all + EH2, rel_all,
                                                prA, prB, rpj, ebuf, EH2, R_REL);
  segsm(ebuf, ebuf, ei_all + EH2, N_ENT, EH2);
  srea_attn_scatter<<<gsz(EH2, 8), TPB, 0, stream>>>(out, 800, ei_all, ei_all + EH2, ebuf,
                                                     out + EHID, 800, EHID, EH2);
  srea_relu2d<<<gsz((long)N_ENT * EHID, TPB), TPB, 0, stream>>>(out + EHID, 800, N_ENT, EHID);

  // ---- relation -> entity GAT on triples (out cols 600..699 and 700..799) ----
  srea_rowdot<<<gsz(N_ENT, 8), TPB, 0, stream>>>(out, 800, g2e_ah, prA, N_ENT, 2 * EHID, 0);
  srea_rowdot<<<gsz(N_ENT, 8), TPB, 0, stream>>>(out, 800, g2e_at, prB, N_ENT, 2 * EHID, 0);
  srea_rowdot<<<gsz(R_REL, 8), TPB, 0, stream>>>(xr, RHID, g2e_ar, arpj, R_REL, RHID, 0);

  // head side
  srea_g2e_e<<<gsz(E_TRI, TPB), TPB, 0, stream>>>(edge_index, rel, prA, arpj, ebuf, E_TRI);
  segsm(ebuf, ebuf, edge_index, N_ENT, E_TRI);
  srea_attn_scatter<<<gsz(E_TRI, 8), TPB, 0, stream>>>(xr, RHID, rel, edge_index, ebuf,
                                                       out + 600, 800, RHID, E_TRI);
  // tail side
  srea_g2e_e<<<gsz(E_TRI, TPB), TPB, 0, stream>>>(edge_index + E_TRI, rel, prB, arpj, ebuf, E_TRI);
  segsm(ebuf, ebuf, edge_index + E_TRI, N_ENT, E_TRI);
  srea_attn_scatter<<<gsz(E_TRI, 8), TPB, 0, stream>>>(xr, RHID, rel, edge_index + E_TRI, ebuf,
                                                       out + 700, 800, RHID, E_TRI);
}
